// DTWMatcher_22600117911560
// MI455X (gfx1250) — matrix operand order repaired and checked by reading, not compiled, not run on hardware
//
#include <hip/hip_runtime.h>
#include <hip/hip_bf16.h>
#include <stdint.h>

// ---------------------------------------------------------------------------
// MI455X (gfx1250, wave32) implementation of DropDTW matcher.
//  - GEMM: v_wmma_f32_16x16x32_bf16 with hi/lo bf16 split (3 WMMAs / K-step)
//  - dropline: exact radix select (11/11/10-bit LDS histograms)
//  - DTW: prefix-min closed form, persistent 32-wave workgroup, P in LDS
// ---------------------------------------------------------------------------

typedef __attribute__((ext_vector_type(16))) __bf16 bf16x16;
typedef __attribute__((ext_vector_type(8)))  __bf16 bf16x8;
typedef __attribute__((ext_vector_type(8)))  float  f32x8;

#define K_TEXT 1024
#define N_EVT  8192
#define DIMS   1024
#define INF_C  1.0e9f
// ascending 0-indexed rank of dropline: numel - max(1, int(numel*0.3))
//   numel = 8388608, k = int(8388608*0.3) = 2516582  ->  rank 5872026
#define RANK_R 5872026u

struct SelState { unsigned r; unsigned prefix; float dropline; float mincost; };

__device__ __forceinline__ unsigned mono_key(float f) {
  unsigned u = __float_as_uint(f);
  return (u & 0x80000000u) ? ~u : (u | 0x80000000u);   // ascending float -> ascending uint
}
__device__ __forceinline__ float key_to_float(unsigned k) {
  unsigned u = (k & 0x80000000u) ? (k & 0x7fffffffu) : ~k;
  return __uint_as_float(u);
}

// --------------------------- normalize + bf16 split -------------------------
__global__ __launch_bounds__(256) void nrm_split_kernel(const float* __restrict__ src,
                                                        __bf16* __restrict__ hi,
                                                        __bf16* __restrict__ lo) {
  const int row = blockIdx.x, t = threadIdx.x, lane = t & 31, wid = t >> 5;
  const float4 v = ((const float4*)(src + (size_t)row * DIMS))[t];   // 256 thr x 4 = 1024
  float ss = v.x * v.x + v.y * v.y + v.z * v.z + v.w * v.w;
#pragma unroll
  for (int d = 16; d >= 1; d >>= 1) ss += __shfl_down(ss, d, 32);
  __shared__ float wpart[8];
  __shared__ float stot;
  if (lane == 0) wpart[wid] = ss;
  __syncthreads();
  if (t == 0) { float s = 0.f; for (int i = 0; i < 8; ++i) s += wpart[i]; stot = s; }
  __syncthreads();
  const float inv = 1.0f / fmaxf(sqrtf(stot), 1e-12f);
  const float e[4] = {v.x * inv, v.y * inv, v.z * inv, v.w * inv};
  const size_t base = (size_t)row * DIMS + (size_t)t * 4;
#pragma unroll
  for (int i = 0; i < 4; ++i) {
    __bf16 h = (__bf16)e[i];
    hi[base + i] = h;
    lo[base + i] = (__bf16)(e[i] - (float)h);          // residual for 3xBF16 f32 emulation
  }
}

// ------------------------------- WMMA GEMM ---------------------------------
__device__ __forceinline__ bf16x16 frag_from(const __bf16* p0, const __bf16* p1) {
  bf16x8 c0 = *(const bf16x8*)p0;
  bf16x8 c1 = *(const bf16x8*)p1;
  bf16x16 r;
#pragma unroll
  for (int i = 0; i < 8; ++i) { r[i] = c0[i]; r[i + 8] = c1[i]; }
  return r;
}

__global__ __launch_bounds__(256) void wmma_gemm_kernel(
    const __bf16* __restrict__ Thi, const __bf16* __restrict__ Tlo,
    const __bf16* __restrict__ Ehi, const __bf16* __restrict__ Elo,
    float* __restrict__ sim) {
  const int t = threadIdx.x, lane = t & 31, w = t >> 5;
  const int lm = lane & 15, h = lane >> 4;
  const int wm = w >> 2, wn = w & 3;                    // 2x4 waves per block
  const int m0 = blockIdx.x * 64 + wm * 32;             // block tile: 64 x 128
  const int n0 = blockIdx.y * 128 + wn * 32;            // wave tile : 32 x 32

  f32x8 acc[2][2] = {};
  const __bf16 *aH[2], *aL[2], *bH[2], *bL[2];
#pragma unroll
  for (int mi = 0; mi < 2; ++mi) {
    size_t ro = (size_t)(m0 + mi * 16 + lm) * DIMS;
    aH[mi] = Thi + ro; aL[mi] = Tlo + ro;
  }
#pragma unroll
  for (int ni = 0; ni < 2; ++ni) {
    size_t ro = (size_t)(n0 + ni * 16 + lm) * DIMS;     // B = E^T: lane = output column
    bH[ni] = Ehi + ro; bL[ni] = Elo + ro;
  }

  for (int kb = 0; kb < DIMS; kb += 32) {
    __builtin_prefetch(aH[0] + kb + 64, 0, 1);          // global_prefetch_b8
    __builtin_prefetch(bH[0] + kb + 64, 0, 1);
    bf16x16 ah[2], al[2], bh[2], bl[2];
#pragma unroll
    for (int mi = 0; mi < 2; ++mi) {
      // A 16x32 bf16 layout: lane-half h -> K chunks {8h..8h+7} and {16+8h..+7}
      ah[mi] = frag_from(aH[mi] + kb + 8 * h, aH[mi] + kb + 16 + 8 * h);
      al[mi] = frag_from(aL[mi] + kb + 8 * h, aL[mi] + kb + 16 + 8 * h);
    }
#pragma unroll
    for (int ni = 0; ni < 2; ++ni) {
      // B 32x16 bf16 layout: lane-half h -> K chunks {8h..8h+7} and {16+8h..+7}
      bh[ni] = frag_from(bH[ni] + kb + 8 * h, bH[ni] + kb + 16 + 8 * h);
      bl[ni] = frag_from(bL[ni] + kb + 8 * h, bL[ni] + kb + 16 + 8 * h);
    }
#pragma unroll
    for (int mi = 0; mi < 2; ++mi)
#pragma unroll
      for (int ni = 0; ni < 2; ++ni) {
        // f32-accurate product via bf16 split: hi*hi + hi*lo + lo*hi
        acc[mi][ni] = __builtin_amdgcn_wmma_f32_16x16x32_bf16(
            false, ah[mi], false, bh[ni], (short)0, acc[mi][ni], false, false);
        acc[mi][ni] = __builtin_amdgcn_wmma_f32_16x16x32_bf16(
            false, ah[mi], false, bl[ni], (short)0, acc[mi][ni], false, false);
        acc[mi][ni] = __builtin_amdgcn_wmma_f32_16x16x32_bf16(
            false, al[mi], false, bh[ni], (short)0, acc[mi][ni], false, false);
      }
  }
  // C/D layout: VGPR v -> M = v + 8*(lane>=16), lane&15 -> N
#pragma unroll
  for (int mi = 0; mi < 2; ++mi)
#pragma unroll
    for (int ni = 0; ni < 2; ++ni) {
      const int col = n0 + ni * 16 + lm;
      const int rb  = m0 + mi * 16 + 8 * h;
#pragma unroll
      for (int v = 0; v < 8; ++v)
        sim[(size_t)(rb + v) * N_EVT + col] = acc[mi][ni][v];
    }
}

// ---------------------- exact radix select for dropline ---------------------
__global__ void init_kernel(unsigned* h0, unsigned* h1, unsigned* h2, SelState* st) {
  int i = blockIdx.x * 256 + threadIdx.x;
  if (i < 2048) { h0[i] = 0; h1[i] = 0; }
  if (i < 1024) h2[i] = 0;
  if (i == 0) { st->r = RANK_R; st->prefix = 0; st->dropline = 0.f; st->mincost = 0.f; }
}

__global__ __launch_bounds__(256) void hist_kernel(const float* __restrict__ sim, unsigned n,
                                                   unsigned* __restrict__ hist,
                                                   const SelState* __restrict__ st, int pass) {
  __shared__ unsigned lh[2048];
  for (int i = threadIdx.x; i < 2048; i += 256) lh[i] = 0;
  __syncthreads();
  const unsigned pfx = st->prefix;
  for (unsigned i = blockIdx.x * 256 + threadIdx.x; i < n; i += gridDim.x * 256) {
    const unsigned key = mono_key(sim[i]);
    if (pass == 0) atomicAdd(&lh[key >> 21], 1u);
    else if (pass == 1) { if ((key >> 21) == pfx) atomicAdd(&lh[(key >> 10) & 2047u], 1u); }
    else               { if ((key >> 10) == pfx) atomicAdd(&lh[key & 1023u], 1u); }
  }
  __syncthreads();
  for (int i = threadIdx.x; i < 2048; i += 256)
    if (lh[i]) atomicAdd(&hist[i], lh[i]);
}

__global__ void scan_kernel(const unsigned* __restrict__ hist, int nbins, SelState* st, int pass) {
  if (threadIdx.x != 0 || blockIdx.x != 0) return;
  unsigned r = st->r, cum = 0;
  int sel = nbins - 1;
  for (int i = 0; i < nbins; ++i) {
    unsigned c = hist[i];
    if (cum + c > r) { sel = i; break; }
    cum += c;
  }
  st->r = r - cum;
  if (pass == 0)      st->prefix = (unsigned)sel;
  else if (pass == 1) st->prefix = (st->prefix << 11) | (unsigned)sel;
  else {
    unsigned key = (st->prefix << 10) | (unsigned)sel;
    st->dropline = key_to_float(key);
  }
}

// ------------------------------- DropDTW DP ---------------------------------
// D0_r[j] = zx[r-1][j-1] + P_{r-1}[j-1]   (P = inclusive prefix-min of D0 row)
// D1_r[j] = exclusive prefix-min of D0_r  (drop costs are zero)
// B[r][j] = (D0 <= D1) : the only info backtracking needs.
__global__ __launch_bounds__(1024) void dtw_kernel(const float* __restrict__ sim,
                                                   SelState* __restrict__ st,
                                                   unsigned char* __restrict__ Bt) {
  __shared__ float P[8200];
  __shared__ float wtot[32];
  const int t = threadIdx.x, lane = t & 31, wid = t >> 5;
  const float dropline = st->dropline;
  for (int i = t; i < 8193; i += 1024) P[i] = 0.0f;                 // row-0 prefix-min
  for (int i = t; i < 8193; i += 1024) Bt[i] = (i == 0) ? 1 : 0;    // row-0 bits
  __syncthreads();

  for (int r = 1; r <= K_TEXT; ++r) {
    const float* z = sim + (size_t)(r - 1) * N_EVT;
    float p[8], d0[8];
#pragma unroll
    for (int e = 0; e < 8; ++e) p[e] = P[t * 8 + e];                // P_{r-1}[j-1]
    const float4* zp = (const float4*)(z + t * 8);
    const float4 z0 = zp[0], z1 = zp[1];
    d0[0] = (dropline - z0.x) + p[0]; d0[1] = (dropline - z0.y) + p[1];
    d0[2] = (dropline - z0.z) + p[2]; d0[3] = (dropline - z0.w) + p[3];
    d0[4] = (dropline - z1.x) + p[4]; d0[5] = (dropline - z1.y) + p[5];
    d0[6] = (dropline - z1.z) + p[6]; d0[7] = (dropline - z1.w) + p[7];

    float pm = d0[0];
#pragma unroll
    for (int e = 1; e < 8; ++e) pm = fminf(pm, d0[e]);
    float inc = pm;                                                  // wave inclusive min-scan
#pragma unroll
    for (int d = 1; d < 32; d <<= 1) {
      float o = __shfl_up(inc, d, 32);
      if (lane >= d) inc = fminf(inc, o);
    }
    if (lane == 31) wtot[wid] = inc;
    __syncthreads();
    if (wid == 0) {                                                  // scan 32 wave totals
      float v = wtot[lane];
#pragma unroll
      for (int d = 1; d < 32; d <<= 1) {
        float o = __shfl_up(v, d, 32);
        if (lane >= d) v = fminf(v, o);
      }
      wtot[lane] = v;
    }
    __syncthreads();
    float laneExcl = __shfl_up(inc, 1, 32);
    if (lane == 0) laneExcl = INF_C;
    float carry = fminf((wid == 0) ? INF_C : wtot[wid - 1], laneExcl);  // D1 at j = t*8+1

    unsigned char* brow = Bt + (size_t)r * 8193;
#pragma unroll
    for (int e = 0; e < 8; ++e) {
      brow[t * 8 + 1 + e] = (d0[e] <= carry) ? (unsigned char)1 : (unsigned char)0;
      carry = fminf(carry, d0[e]);
      P[t * 8 + 1 + e] = carry;                                      // new inclusive prefix-min
    }
    if (t == 0) { P[0] = INF_C; brow[0] = 1; }                       // D0_r[0] = INF (r>=1)
    __syncthreads();
  }
  if (t == 0) st->mincost = P[8192];   // = min(D0[K,N], D1[K,N])
}

// -------------------------------- backtrack ---------------------------------
__global__ void backtrack_kernel(const unsigned char* __restrict__ Bt, int* __restrict__ q) {
  if (threadIdx.x != 0 || blockIdx.x != 0) return;
  for (int i = 0; i < K_TEXT; ++i) q[i] = 0;
  int zi = K_TEXT, j = N_EVT;
  int s = Bt[(size_t)zi * 8193 + j] ? 0 : 1;
  for (int it = 0; it < K_TEXT + N_EVT; ++it) {
    const bool active = zi > 0;
    const bool match = active && (s == 0);
    if (match) q[zi - 1] = j - 1;
    const int nzi = match ? zi - 1 : zi;
    const int nj = active ? j - 1 : j;
    const int cj = nj < 0 ? 0 : nj;
    const int ns = Bt[(size_t)nzi * 8193 + cj] ? 0 : 1;
    if (active) s = ns;
    zi = nzi; j = nj;
  }
}

__global__ void finalize_kernel(const int* __restrict__ q, const SelState* __restrict__ st,
                                float* __restrict__ out, int out_size) {
  const int i = blockIdx.x * 256 + threadIdx.x;
  if (i >= out_size) return;
  if (i < K_TEXT) out[i] = (float)q[i];
  else if (i < 2 * K_TEXT) out[i] = (float)(i - K_TEXT);
  else out[i] = st->mincost;
}

// ------------------------------- launcher -----------------------------------
extern "C" void kernel_launch(void* const* d_in, const int* in_sizes, int n_in,
                              void* d_out, int out_size, void* d_ws, size_t ws_size,
                              hipStream_t stream) {
  const float* text = (const float*)d_in[0];   // [1024, 1024] f32
  const float* evt  = (const float*)d_in[1];   // [8192, 1024] f32

  char* w = (char*)d_ws;
  auto take = [&](size_t bytes) { char* p = w; w += (bytes + 255) & ~(size_t)255; return p; };
  __bf16* Thi = (__bf16*)take((size_t)2 * K_TEXT * DIMS);
  __bf16* Tlo = (__bf16*)take((size_t)2 * K_TEXT * DIMS);
  __bf16* Ehi = (__bf16*)take((size_t)2 * N_EVT * DIMS);
  __bf16* Elo = (__bf16*)take((size_t)2 * N_EVT * DIMS);
  float*  sim = (float*)take((size_t)4 * K_TEXT * N_EVT);
  unsigned* h0 = (unsigned*)take(2048 * 4);
  unsigned* h1 = (unsigned*)take(2048 * 4);
  unsigned* h2 = (unsigned*)take(1024 * 4);
  SelState* st = (SelState*)take(256);
  int* q = (int*)take(K_TEXT * 4);
  unsigned char* Bt = (unsigned char*)take((size_t)(K_TEXT + 1) * 8193);

  nrm_split_kernel<<<K_TEXT, 256, 0, stream>>>(text, Thi, Tlo);
  nrm_split_kernel<<<N_EVT, 256, 0, stream>>>(evt, Ehi, Elo);
  wmma_gemm_kernel<<<dim3(16, 64), 256, 0, stream>>>(Thi, Tlo, Ehi, Elo, sim);

  const unsigned n = (unsigned)K_TEXT * (unsigned)N_EVT;
  init_kernel<<<8, 256, 0, stream>>>(h0, h1, h2, st);
  hist_kernel<<<256, 256, 0, stream>>>(sim, n, h0, st, 0);
  scan_kernel<<<1, 1, 0, stream>>>(h0, 2048, st, 0);
  hist_kernel<<<256, 256, 0, stream>>>(sim, n, h1, st, 1);
  scan_kernel<<<1, 1, 0, stream>>>(h1, 2048, st, 1);
  hist_kernel<<<256, 256, 0, stream>>>(sim, n, h2, st, 2);
  scan_kernel<<<1, 1, 0, stream>>>(h2, 1024, st, 2);

  dtw_kernel<<<1, 1024, 0, stream>>>(sim, st, Bt);
  backtrack_kernel<<<1, 1, 0, stream>>>(Bt, q);
  finalize_kernel<<<(out_size + 255) / 256, 256, 0, stream>>>(q, st, (float*)d_out, out_size);
}
